// RadialAttention_84945863180471
// MI455X (gfx1250) — matrix operand order repaired and checked by reading, not compiled, not run on hardware
//
#include <hip/hip_runtime.h>

// ---------------------------------------------------------------------------
// RadialAttention for MI455X (gfx1250, wave32, WMMA bf16 16x16x32, f32 acc)
// B=8, C=256, H=W=64, N=4096, RB=8
// Round 2: async-to-LDS double-buffered K staging in the attention kernel
// (global_load_async_to_lds_b128 + s_wait_asynccnt, per CDNA5 ISA).
// ---------------------------------------------------------------------------

typedef __bf16 bf16;
typedef __attribute__((ext_vector_type(16))) __bf16 v16bf;
typedef __attribute__((ext_vector_type(8)))  __bf16 v8bf;
typedef __attribute__((ext_vector_type(4)))  __bf16 v4bf;
typedef __attribute__((ext_vector_type(8)))  float  v8f;

static constexpr int cB = 8;
static constexpr int cC = 256;
static constexpr int cN = 4096;          // 64*64 tokens
static constexpr float cScale = 0.0625f; // 1/sqrt(256)

union ABu { v16bf v; v8bf h[2]; };

// D = A(16x32) * B(32x16) + C, bf16 inputs, f32 accumulate
__device__ __forceinline__ v8f wmma_bf16(v16bf a, v16bf b, v8f c) {
    return __builtin_amdgcn_wmma_f32_16x16x32_bf16(
        /*neg_a=*/false, a, /*neg_b=*/false, b,
        /*c_mod=*/(short)0, c, /*reuse_a=*/false, /*reuse_b=*/false);
}

// A fragment: 16x32, row-major source, row m contiguous in k.
// lane: m = lane&15; half h = lane>>4 owns k chunks [h*8..h*8+7], [16+h*8..]
__device__ __forceinline__ v16bf ld_afrag(const bf16* p, int ld, int lane) {
    const int m = lane & 15, h = lane >> 4;
    ABu u;
    u.h[0] = *(const v8bf*)(p + (size_t)m * ld + h * 8);
    u.h[1] = *(const v8bf*)(p + (size_t)m * ld + 16 + h * 8);
    return u.v;
}

// B fragment: 32x16 (KxN), source stored n-major with k contiguous:
// element (k,n) at p + n*ld + k. lane: n = lane&15; half h owns k chunks [h*8..h*8+7], [16+h*8..]
__device__ __forceinline__ v16bf ld_bfrag(const bf16* p, int ld, int lane) {
    const int n = lane & 15, h = lane >> 4;
    ABu u;
    u.h[0] = *(const v8bf*)(p + (size_t)n * ld + h * 8);
    u.h[1] = *(const v8bf*)(p + (size_t)n * ld + 16 + h * 8);
    return u.v;
}

// CDNA5 async copy: 16 bytes global -> LDS per lane, tracked by ASYNCcnt.
// Generic pointers to __shared__ carry the LDS offset in their low 32 bits
// (ISA 10.2 aperture rules), which is what VDST wants.
__device__ __forceinline__ void async_ld16(void* lds, const void* g) {
    asm volatile("global_load_async_to_lds_b128 %0, %1, off"
                 :: "v"((unsigned)(size_t)lds), "v"(g)
                 : "memory");
}
#define WAIT_ASYNC(n) asm volatile("s_wait_asynccnt %0" :: "i"(n) : "memory")

// radial bin scale for a flat token index; ras = 8-entry table (in LDS)
__device__ __forceinline__ float radial_r(const float* ras, int tok) {
    const int y = tok >> 6, x = tok & 63;
    const float dy = (float)(y - 32), dx = (float)(x - 32);
    const float rad = __fsqrt_rn(dx * dx + dy * dy);
    int idx = (int)(rad * (7.0f / 45.25483399593904f)); // 7/sqrt(32^2+32^2)
    idx = idx < 0 ? 0 : (idx > 7 ? 7 : idx);
    return ras[idx];
}

// ---------------------------------------------------------------------------
// Kernel 1: convert the four CxC fp32 weight matrices to bf16
// ---------------------------------------------------------------------------
__global__ __launch_bounds__(256) void cvtw_kernel(
    const float* __restrict__ Wq, const float* __restrict__ Wk,
    const float* __restrict__ Wv, const float* __restrict__ Wo,
    bf16* __restrict__ wbf) {
    const int idx = blockIdx.x * 256 + threadIdx.x;   // 65536 threads
    const int mat = idx >> 14;                         // 16384 threads / matrix
    const int off = (idx & 16383) * 4;
    const float* src = (mat == 0) ? Wq : (mat == 1) ? Wk : (mat == 2) ? Wv : Wo;
    const float4 f = *(const float4*)(src + off);
    v4bf o;
    o[0] = (bf16)f.x; o[1] = (bf16)f.y; o[2] = (bf16)f.z; o[3] = (bf16)f.w;
    *(v4bf*)(wbf + (size_t)mat * cC * cC + off) = o;
}

// ---------------------------------------------------------------------------
// Kernel 2: x [b][c][n] f32 -> xT [b][n][c] bf16 via LDS tile transpose
// ---------------------------------------------------------------------------
__global__ __launch_bounds__(256) void xT_kernel(
    const float* __restrict__ x, bf16* __restrict__ xT) {
    __shared__ bf16 tile[64 * 256];
    const int tid = threadIdx.x;
    const int b = blockIdx.y;
    const int tb = blockIdx.x * 64;
    const float* xb = x + (size_t)b * cC * cN;
    for (int i = tid; i < 64 * 256 / 4; i += 256) {
        const int c  = i >> 4;
        const int t4 = (i & 15) * 4;
        const float4 f = *(const float4*)(xb + (size_t)c * cN + tb + t4);
        tile[(t4 + 0) * 256 + c] = (bf16)f.x;
        tile[(t4 + 1) * 256 + c] = (bf16)f.y;
        tile[(t4 + 2) * 256 + c] = (bf16)f.z;
        tile[(t4 + 3) * 256 + c] = (bf16)f.w;
    }
    __syncthreads();
    bf16* xo = xT + ((size_t)b * cN + tb) * cC;
    for (int i = tid; i < 64 * 256 / 8; i += 256)
        *(v8bf*)(xo + i * 8) = *(const v8bf*)(tile + i * 8);
}

// ---------------------------------------------------------------------------
// Kernel 3: fused QKV projection (A = W rows per q/k/v; xT B-frags shared across q/k/v).
//   q,k token-major [n][c]; v stored channel-major [c][n] for P.V B-frags.
// grid (C/64, N/64, B), 128 threads; wave = 16 out-channels x 64 tokens.
// ---------------------------------------------------------------------------
__global__ __launch_bounds__(128) void qkv_kernel(
    const bf16* __restrict__ wbf, const bf16* __restrict__ xT,
    bf16* __restrict__ q, bf16* __restrict__ k, bf16* __restrict__ vT) {
    const int tid = threadIdx.x, lane = tid & 31, w = tid >> 5;
    const int m15 = lane & 15, hh = lane >> 4;
    const int b = blockIdx.z;
    const int ow = blockIdx.x * 64 + w * 16;
    const int tb = blockIdx.y * 64;
    const bf16* xb = xT + ((size_t)b * cN + tb) * cC;

    v8f acc[3][4] = {};
#pragma unroll
    for (int ci = 0; ci < 8; ++ci) {
        v16bf bfr[4];
#pragma unroll
        for (int s = 0; s < 4; ++s)
            bfr[s] = ld_bfrag(xb + (size_t)(s * 16) * cC + ci * 32, cC, lane);
#pragma unroll
        for (int y = 0; y < 3; ++y) {
            const v16bf a = ld_afrag(
                wbf + (size_t)y * cC * cC + (size_t)ow * cC + ci * 32, cC, lane);
#pragma unroll
            for (int s = 0; s < 4; ++s)
                acc[y][s] = wmma_bf16(a, bfr[s], acc[y][s]);
        }
    }

    bf16* outs01[2] = {q, k};
#pragma unroll
    for (int y = 0; y < 2; ++y) {
        bf16* yb = outs01[y] + (size_t)b * cN * cC;
#pragma unroll
        for (int s = 0; s < 4; ++s) {
            const int token = tb + s * 16 + m15;
            v8bf pk;
#pragma unroll
            for (int r = 0; r < 8; ++r) pk[r] = (bf16)acc[y][s][r];
            *(v8bf*)(yb + (size_t)token * cC + ow + hh * 8) = pk;
        }
    }
    bf16* vb = vT + (size_t)b * cC * cN;
#pragma unroll
    for (int s = 0; s < 4; ++s) {
        const int token = tb + s * 16 + m15;
#pragma unroll
        for (int r = 0; r < 8; ++r)
            vb[(size_t)(ow + hh * 8 + r) * cN + token] = (bf16)acc[2][s][r];
    }
}

// ---------------------------------------------------------------------------
// Kernel 4: flash attention over 64-row tiles.
// grid (N/64, B), 128 threads (4 waves); wave owns 16 query rows.
// Q tile + double-buffered K tiles staged via async-to-LDS (ASYNCcnt);
// the copy of K tile j+1 overlaps the WMMA/softmax work of tile j.
// ---------------------------------------------------------------------------
__global__ __launch_bounds__(128) void attn_kernel(
    const bf16* __restrict__ q, const bf16* __restrict__ k,
    const bf16* __restrict__ vT, const float* __restrict__ ra,
    bf16* __restrict__ o) {
    __shared__ bf16 Qs[64 * 256];        // 32 KB query tile
    __shared__ bf16 Ks[2][64 * 256];     // 64 KB double-buffered K tiles
    __shared__ bf16 Ps[4][16 * 64];      // 8 KB  per-wave P tiles
    __shared__ float ras[8];

    const int tid = threadIdx.x, lane = tid & 31, w = tid >> 5;
    const int m15 = lane & 15, hh = lane >> 4;
    const int b = blockIdx.y;
    const int rowbase = blockIdx.x * 64;

    const bf16* qb = q  + ((size_t)b * cN + rowbase) * cC;
    const bf16* kb = k  + (size_t)b * cN * cC;
    const bf16* vb = vT + (size_t)b * cC * cN;

    if (tid < 8) ras[tid] = ra[tid];

    // ---- async stage: Q tile (contiguous 32 KB) + K tile 0 ----
#pragma unroll
    for (int i = 0; i < 16; ++i) {
        const int c = i * 128 + tid;                 // 16B chunk id
        async_ld16(Qs + c * 8, qb + c * 8);
    }
#pragma unroll
    for (int i = 0; i < 16; ++i) {
        const int c = i * 128 + tid;
        async_ld16(&Ks[0][0] + c * 8, kb + c * 8);   // tile 0 = rows 0..63
    }
    WAIT_ASYNC(0);
    __syncthreads();

    float rr[8];
#pragma unroll
    for (int r = 0; r < 8; ++r)
        rr[r] = cScale * radial_r(ras, rowbase + w * 16 + hh * 8 + r);

    float mrow[8], lrow[8];
#pragma unroll
    for (int r = 0; r < 8; ++r) { mrow[r] = -3.0e38f; lrow[r] = 0.0f; }
    v8f oacc[16] = {};

    const bf16* QsW = Qs + (size_t)(w * 16) * 256;
    bf16* pw = Ps[w];

    for (int j = 0; j < 64; ++j) {
        const int jb = j * 64;
        const bf16* kcur = &Ks[j & 1][0];

        // kick off async copy of K tile j+1 into the other buffer
        if (j + 1 < 64) {
            bf16* knext = &Ks[(j + 1) & 1][0];
            const bf16* ksrc = kb + (size_t)(jb + 64) * cC;
#pragma unroll
            for (int i = 0; i < 16; ++i) {
                const int c = i * 128 + tid;
                async_ld16(knext + c * 8, ksrc + c * 8);
            }
        }

        // ---- S = Q.K^T from LDS tiles ----
        v8f s[4] = {};
#pragma unroll
        for (int ci = 0; ci < 8; ++ci) {
            const v16bf a = ld_afrag(QsW + ci * 32, 256, lane);
#pragma unroll
            for (int t = 0; t < 4; ++t) {
                const v16bf bk =
                    ld_bfrag(kcur + (size_t)(t * 16) * 256 + ci * 32, 256, lane);
                s[t] = wmma_bf16(a, bk, s[t]);
            }
        }

        // ---- radial bias + online softmax ----
        float rc[4];
#pragma unroll
        for (int t = 0; t < 4; ++t) rc[t] = radial_r(ras, jb + t * 16 + m15);
#pragma unroll
        for (int t = 0; t < 4; ++t)
#pragma unroll
            for (int r = 0; r < 8; ++r) s[t][r] *= rr[r] * rc[t];

        float alpha[8];
#pragma unroll
        for (int r = 0; r < 8; ++r) {
            float vm = fmaxf(fmaxf(s[0][r], s[1][r]), fmaxf(s[2][r], s[3][r]));
#pragma unroll
            for (int msk = 1; msk < 16; msk <<= 1)
                vm = fmaxf(vm, __shfl_xor(vm, msk, 32));
            const float mnew = fmaxf(mrow[r], vm);
            alpha[r] = __expf(mrow[r] - mnew);
            mrow[r] = mnew;
        }

        float ls[8];
#pragma unroll
        for (int r = 0; r < 8; ++r) ls[r] = 0.0f;
#pragma unroll
        for (int t = 0; t < 4; ++t)
#pragma unroll
            for (int r = 0; r < 8; ++r) {
                const float p = __expf(s[t][r] - mrow[r]);
                ls[r] += p;
                pw[(hh * 8 + r) * 64 + t * 16 + m15] = (bf16)p;
            }
#pragma unroll
        for (int r = 0; r < 8; ++r) {
            float lsum = ls[r];
#pragma unroll
            for (int msk = 1; msk < 16; msk <<= 1)
                lsum += __shfl_xor(lsum, msk, 32);
            lrow[r] = lrow[r] * alpha[r] + lsum;
        }
#pragma unroll
        for (int cf = 0; cf < 16; ++cf)
#pragma unroll
            for (int r = 0; r < 8; ++r) oacc[cf][r] *= alpha[r];

        // ---- O += P.V : A = P (LDS bounce), B = vT (m contiguous) ----
        const v16bf pa0 = ld_afrag(pw, 64, lane);
        const v16bf pa1 = ld_afrag(pw + 32, 64, lane);
        if (j + 1 < 64)
            __builtin_prefetch(vb + jb + 64, 0, 1);
#pragma unroll
        for (int cf = 0; cf < 16; ++cf) {
            const v16bf bv0 =
                ld_bfrag(vb + (size_t)(cf * 16) * cN + jb, cN, lane);
            const v16bf bv1 =
                ld_bfrag(vb + (size_t)(cf * 16) * cN + jb + 32, cN, lane);
            oacc[cf] = wmma_bf16(pa0, bv0, oacc[cf]);
            oacc[cf] = wmma_bf16(pa1, bv1, oacc[cf]);
        }

        // K tile j+1 must be fully in LDS before the next iteration reads it
        WAIT_ASYNC(0);
        __syncthreads();
    }

    // ---- finalize: O /= l, store bf16 token-major [n][c] ----
    float inv[8];
#pragma unroll
    for (int r = 0; r < 8; ++r) inv[r] = 1.0f / lrow[r];
    bf16* ob = o + ((size_t)b * cN + rowbase + w * 16) * cC;
#pragma unroll
    for (int cf = 0; cf < 16; ++cf)
#pragma unroll
        for (int r = 0; r < 8; ++r)
            ob[(size_t)(hh * 8 + r) * cC + cf * 16 + m15] =
                (bf16)(oacc[cf][r] * inv[r]);
}

// ---------------------------------------------------------------------------
// Kernel 5: output projection + residual, stores [B][C][H][W] fp32.
// ---------------------------------------------------------------------------
__global__ __launch_bounds__(128) void oproj_kernel(
    const bf16* __restrict__ wo, const bf16* __restrict__ ob,
    const float* __restrict__ x, float* __restrict__ out) {
    const int tid = threadIdx.x, lane = tid & 31, w = tid >> 5;
    const int m15 = lane & 15, hh = lane >> 4;
    const int b = blockIdx.z;
    const int ow = blockIdx.x * 64 + w * 16;
    const int tb = blockIdx.y * 64;
    const bf16* obb = ob + ((size_t)b * cN + tb) * cC;

    v8f acc[4] = {};
#pragma unroll
    for (int ci = 0; ci < 8; ++ci) {
        const v16bf a = ld_afrag(wo + (size_t)ow * cC + ci * 32, cC, lane);
#pragma unroll
        for (int s = 0; s < 4; ++s) {
            const v16bf bfr =
                ld_bfrag(obb + (size_t)(s * 16) * cC + ci * 32, cC, lane);
            acc[s] = wmma_bf16(a, bfr, acc[s]);
        }
    }
#pragma unroll
    for (int s = 0; s < 4; ++s) {
        const int token = tb + s * 16 + m15;
#pragma unroll
        for (int r = 0; r < 8; ++r) {
            const size_t addr =
                ((size_t)b * cC + ow + hh * 8 + r) * (size_t)cN + token;
            out[addr] = x[addr] + acc[s][r];
        }
    }
}

// ---------------------------------------------------------------------------
extern "C" void kernel_launch(void* const* d_in, const int* in_sizes, int n_in,
                              void* d_out, int out_size, void* d_ws,
                              size_t ws_size, hipStream_t stream) {
    const float* x  = (const float*)d_in[0];
    const float* Wq = (const float*)d_in[1];
    const float* Wk = (const float*)d_in[2];
    const float* Wv = (const float*)d_in[3];
    const float* Wo = (const float*)d_in[4];
    const float* ra = (const float*)d_in[5];
    float* out = (float*)d_out;

    // workspace layout (bf16 elements): weights | xT | q | k | vT | O
    bf16* wbf = (bf16*)d_ws;
    const size_t per = (size_t)cB * cN * cC;        // 8Mi elems per tensor
    bf16* xT = wbf + (size_t)4 * cC * cC;
    bf16* q  = xT + per;
    bf16* k  = q + per;
    bf16* vT = k + per;
    bf16* ob = vT + per;

    cvtw_kernel<<<dim3(256), 256, 0, stream>>>(Wq, Wk, Wv, Wo, wbf);
    xT_kernel<<<dim3(cN / 64, cB), 256, 0, stream>>>(x, xT);
    qkv_kernel<<<dim3(cC / 64, cN / 64, cB), 128, 0, stream>>>(wbf, xT, q, k, vT);
    attn_kernel<<<dim3(cN / 64, cB), 128, 0, stream>>>(q, k, vT, ra, ob);
    oproj_kernel<<<dim3(cC / 64, cN / 64, cB), 128, 0, stream>>>(
        wbf + (size_t)3 * cC * cC, ob, x, out);
}
